// SpatialClustering_82806969467036
// MI455X (gfx1250) — compile-verified, matrix operand order checked by reading, not run on hardware
//
#include <hip/hip_runtime.h>
#include <hip/hip_bf16.h>
#include <stdint.h>

// CDNA5 / gfx1250: wave32, WMMA 16x16x32 f16->f32 used for exact integer
// cross-lane sum reductions inside the clustering loop.

typedef __attribute__((ext_vector_type(16))) _Float16 v16h;
typedef __attribute__((ext_vector_type(8)))  float    v8f;

#define THRESHOLD_F 0.5f
#define LN2_F 0.6931471805599453f
#define MIN_PIXELS_F 64.0f

constexpr int B_  = 4;
constexpr int H_  = 512;
constexpr int W_  = 1024;
constexpr int N_  = H_ * W_;       // 524288 pixels per image
constexpr int NW_ = N_ / 32;       // 16384 bitmask words per image
constexpr int TPB = 1024;          // 32 waves per block
constexpr int WPT = NW_ / TPB;     // 16 words per thread

// Exact wave32 sum reduction: one v_wmma_f32_16x16x32_f16 folds lane L with
// lane L+16 (A = ones 16x32, B slot0 = per-lane value, f32 accumulate), then
// 4 xor-shuffles finish the 16 columns. Inputs are integer counts <= 512, so
// the f16 conversion is exact; accumulation is f32 so totals are exact.
// Call sites are fully uniform (EXEC must be all ones for WMMA).
__device__ __forceinline__ float wave_sum32(float v) {
  v16h a, b;
#pragma unroll
  for (int i = 0; i < 16; ++i) { a[i] = (_Float16)1.0f; b[i] = (_Float16)0.0f; }
  b[0] = (_Float16)v;
  v8f c = {};
  c = __builtin_amdgcn_wmma_f32_16x16x32_f16(false, a, false, b, (short)0, c,
                                             false, false);
  float r = c[0];
  r += __shfl_xor(r, 1, 32);
  r += __shfl_xor(r, 2, 32);
  r += __shfl_xor(r, 4, 32);
  r += __shfl_xor(r, 8, 32);
  return r;  // identical in all lanes
}

// Pass 1: elementwise precompute (bandwidth bound; whole problem is L2
// resident on MI455X: 40MB << 192MB L2). Builds spatial embeddings, seed
// sigmoid, 1/(2*sigma^2), and valid/unclustered bitmasks via wave32 ballot.
__global__ void cluster_init_kernel(const float* __restrict__ tag,
                                    float* __restrict__ ex, float* __restrict__ ey,
                                    float* __restrict__ sa, float* __restrict__ sb,
                                    float* __restrict__ sm,
                                    uint32_t* __restrict__ vmask,
                                    uint32_t* __restrict__ umask,
                                    float* __restrict__ out) {
  const int t = blockIdx.x * blockDim.x + threadIdx.x;  // exact launch: B_*N_
  const int b = t / N_;
  const int p = t - b * N_;
  const int y = p / W_;
  const int x = p - y * W_;
  const float* img = tag + (size_t)b * 5 * N_;
  const float ox = img[(size_t)0 * N_ + p];
  const float oy = img[(size_t)1 * N_ + p];
  const float sx = img[(size_t)2 * N_ + p];
  const float sy = img[(size_t)3 * N_ + p];
  const float sr = img[(size_t)4 * N_ + p];

  float gx, gy;
  if (W_ >= H_) {
    gx = ((float)W_ / (float)H_) * ((float)x / (float)(W_ - 1));
    gy = (float)y / (float)(H_ - 1);
  } else {
    gx = (float)x / (float)(W_ - 1);
    gy = ((float)H_ / (float)W_) * ((float)y / (float)(H_ - 1));
  }

  ex[t] = tanhf(ox) + gx;
  ey[t] = tanhf(oy) + gy;
  sa[t] = 1.0f / (2.0f * sx * sx);  // coefficient for (ex-cx)^2
  sb[t] = 1.0f / (2.0f * sy * sy);
  sm[t] = 1.0f / (1.0f + expf(-sr));  // sigmoid seed
  out[t] = 0.0f;                      // background label 0 every launch

  const bool valid = sr > THRESHOLD_F;
  const unsigned long long bm = __ballot(valid);  // wave32: low 32 bits
  if ((threadIdx.x & 31) == 0) {
    vmask[t >> 5] = (uint32_t)bm;
    umask[t >> 5] = (uint32_t)bm;
  }
}

// Pass 2: one workgroup per image runs the sequential clustering loop.
// Phases per iteration (all barrier-separated, uniform control flow):
//   A) count unclustered + argmax(seed | unclustered)      [wmma reduce]
//   B) Gaussian proposal mask (d2 < ln2), pc / uc counts    [wmma reduce]
//   C) if accepted: label pixels, clear bits, count++
__global__ __launch_bounds__(TPB, 1) void cluster_iter_kernel(
    const float* __restrict__ ex, const float* __restrict__ ey,
    const float* __restrict__ sa, const float* __restrict__ sb,
    const float* __restrict__ sm, const uint32_t* __restrict__ vmask,
    uint32_t* __restrict__ umask, float* __restrict__ out) {
  __shared__ float s_sum[32];
  __shared__ float s_aux[32];
  __shared__ int   s_idx[32];
  __shared__ float s_b[6];  // 0: stop, 1: accept, 2..5: cx, cy, ca, cb

  const int tid  = threadIdx.x;
  const int lane = tid & 31;
  const int wid  = tid >> 5;
  const int b    = blockIdx.x;
  const size_t ib = (size_t)b * N_;
  uint32_t* uncl        = umask + (size_t)b * NW_;
  const uint32_t* vld   = vmask + (size_t)b * NW_;

  int count = 1;  // instance ids start at 1 (0 = background)

  for (;;) {
    // ---- Phase A: popcount(unclustered) + argmax(seed among unclustered) --
    float cnt  = 0.0f;
    float best = 0.0f;  // score = seed where unclustered else 0 -> max=0 ends
    int   bidx = 0;
#pragma unroll 4
    for (int w = 0; w < WPT; ++w) {
      const int wi = w * TPB + tid;
      uint32_t m = uncl[wi];
      cnt += (float)__popc(m);
      const int p0 = wi << 5;
      __builtin_prefetch(sm + ib + p0, 0, 3);  // global_prefetch_b8
      while (m) {
        const int bit = __builtin_ctz(m);
        m &= m - 1;
        const int p = p0 + bit;
        const float s = sm[ib + p];
        if (s > best || (s == best && p < bidx)) { best = s; bidx = p; }
      }
    }
    cnt = wave_sum32(cnt);
#pragma unroll
    for (int mask = 16; mask >= 1; mask >>= 1) {
      const float ov = __shfl_xor(best, mask, 32);
      const int   oi = __shfl_xor(bidx, mask, 32);
      if (ov > best || (ov == best && oi < bidx)) { best = ov; bidx = oi; }
    }
    if (lane == 0) { s_sum[wid] = cnt; s_aux[wid] = best; s_idx[wid] = bidx; }
    __syncthreads();

    if (tid == 0) {
      float tot = 0.0f;
      float bv = s_aux[0];
      int   bi = s_idx[0];
      for (int j = 0; j < 32; ++j) {
        tot += s_sum[j];
        const float v = s_aux[j];
        const int   i = s_idx[j];
        if (v > bv || (v == bv && i < bi)) { bv = v; bi = i; }
      }
      float stop = 0.0f;
      if (!(tot > MIN_PIXELS_F)) {
        stop = 1.0f;                 // while-cond: sum(unclustered) > 64
      } else if (bv < THRESHOLD_F) {
        stop = 1.0f;                 // new_done: best seed below threshold
      } else {
        uncl[bi >> 5] &= ~(1u << (bi & 31));  // remove center from unclustered
        s_b[2] = ex[ib + bi];
        s_b[3] = ey[ib + bi];
        s_b[4] = sa[ib + bi];
        s_b[5] = sb[ib + bi];
      }
      s_b[0] = stop;
    }
    __syncthreads();
    if (s_b[0] != 0.0f) break;  // uniform exit

    const float cx = s_b[2], cy = s_b[3], ca = s_b[4], cb = s_b[5];

    // ---- Phase B: proposal mask + pc/uc counts ----------------------------
    // propw stays register-resident (compile-time indices in unrolled body);
    // the unclustered words are NOT kept live -> Phase C reloads them (no
    // writes occur in between), which removes the VGPR spills seen before.
    float pcf = 0.0f, ucf = 0.0f;
    uint32_t propw[WPT];
#pragma unroll 4
    for (int w = 0; w < WPT; ++w) {
      const int wi = w * TPB + tid;
      const uint32_t uw = uncl[wi];  // reload: center bit may be cleared here
      const uint32_t vw = vld[wi];
      const int p0 = wi << 5;
      const float4* ex4 = reinterpret_cast<const float4*>(ex + ib + p0);
      const float4* ey4 = reinterpret_cast<const float4*>(ey + ib + p0);
      uint32_t prop = 0u;
#pragma unroll
      for (int ck = 0; ck < 8; ++ck) {
        const float4 vx = ex4[ck];
        const float4 vy = ey4[ck];
        float dx, dy, d2;
        dx = vx.x - cx; dy = vy.x - cy; d2 = ca * dx * dx + cb * dy * dy;
        if (d2 < LN2_F) prop |= (1u << (ck * 4 + 0));
        dx = vx.y - cx; dy = vy.y - cy; d2 = ca * dx * dx + cb * dy * dy;
        if (d2 < LN2_F) prop |= (1u << (ck * 4 + 1));
        dx = vx.z - cx; dy = vy.z - cy; d2 = ca * dx * dx + cb * dy * dy;
        if (d2 < LN2_F) prop |= (1u << (ck * 4 + 2));
        dx = vx.w - cx; dy = vy.w - cy; d2 = ca * dx * dx + cb * dy * dy;
        if (d2 < LN2_F) prop |= (1u << (ck * 4 + 3));
      }
      prop &= vw;  // (dist > MARGIN) & valid
      propw[w] = prop;
      pcf += (float)__popc(prop);
      ucf += (float)__popc(prop & uw);
    }
    pcf = wave_sum32(pcf);
    ucf = wave_sum32(ucf);
    if (lane == 0) { s_sum[wid] = pcf; s_aux[wid] = ucf; }
    __syncthreads();
    if (tid == 0) {
      float pc = 0.0f, uc = 0.0f;
      for (int j = 0; j < 32; ++j) { pc += s_sum[j]; uc += s_aux[j]; }
      s_b[1] = ((pc > MIN_PIXELS_F) && (2.0f * uc > pc)) ? 1.0f : 0.0f;
    }
    __syncthreads();

    // ---- Phase C: commit accepted cluster ---------------------------------
    if (s_b[1] != 0.0f) {
      const float fc = (float)count;
#pragma unroll 4
      for (int w = 0; w < WPT; ++w) {
        const uint32_t prop = propw[w];
        if (prop) {
          const int wi = w * TPB + tid;
          uncl[wi] = uncl[wi] & ~prop;  // exact reload: unchanged since B
          const int p0 = wi << 5;
          uint32_t m = prop;
          while (m) {
            const int bit = __builtin_ctz(m);
            m &= m - 1;
            out[ib + p0 + bit] = fc;
          }
        }
      }
      ++count;  // uniform across all threads
    }
    __syncthreads();  // state committed before next Phase A scan
  }
}

extern "C" void kernel_launch(void* const* d_in, const int* in_sizes, int n_in,
                              void* d_out, int out_size, void* d_ws, size_t ws_size,
                              hipStream_t stream) {
  (void)in_sizes; (void)n_in; (void)out_size; (void)ws_size;
  const float* tag = (const float*)d_in[0];
  float* out = (float*)d_out;

  const size_t T = (size_t)B_ * N_;
  float* ex = (float*)d_ws;
  float* ey = ex + T;
  float* sa = ey + T;
  float* sb = sa + T;
  float* sm = sb + T;
  uint32_t* vmask = (uint32_t*)(sm + T);
  uint32_t* umask = vmask + (T / 32);

  cluster_init_kernel<<<dim3((unsigned)(T / 256)), dim3(256), 0, stream>>>(
      tag, ex, ey, sa, sb, sm, vmask, umask, out);
  cluster_iter_kernel<<<dim3(B_), dim3(TPB), 0, stream>>>(
      ex, ey, sa, sb, sm, vmask, umask, out);
}
